// DynamicDepthSeparableConv1dTemplateAttention_9414568313153
// MI455X (gfx1250) — matrix operand order repaired and checked by reading, not compiled, not run on hardware
//
#include <hip/hip_runtime.h>
#include <hip/hip_bf16.h>

typedef __attribute__((ext_vector_type(16))) _Float16 v16h;
typedef __attribute__((ext_vector_type(8)))  _Float16 v8h;
typedef __attribute__((ext_vector_type(8)))  float    v8f;

#define B_   16
#define QKC_ 32
#define H_   8
#define L_   1024
#define NH_  (B_ * H_)   // 128 heads

// ---------------------------------------------------------------------------
// Encoder for Q / K: pointwise (32->256) then gated depthwise k=3 / k=15.
// Output layout: f16 out[(b*8+h)*1024 + l][c] with o = h*32 + c  (row = 32 f16,
// 64B — exactly what the WMMA A/B per-lane loads want).
// Grid: (16 b, 16 ochunks of 16, 4 ltiles of 256), block 256.
// ---------------------------------------------------------------------------
__global__ void __launch_bounds__(256)
encode_qk_kernel(const float* __restrict__ x,      // [16][32][1024]
                 const float* __restrict__ pw,     // [256][32]
                 const float* __restrict__ dw3,    // [256][3]
                 const float* __restrict__ dw15,   // [256][15]
                 const float* __restrict__ gate,   // [2]
                 float scale,
                 _Float16* __restrict__ out)       // [128][1024][32]
{
    __shared__ float xs[32][272];    // halo'd input, j -> l = l0 - 7 + j
    __shared__ float ps[16][272];    // pointwise result for this o-chunk
    __shared__ float pw_s[16][32];
    __shared__ float dw3_s[16][3];
    __shared__ float dw15_s[16][15];

    const int tid = threadIdx.x;
    const int b   = blockIdx.x;
    const int oc0 = blockIdx.y * 16;
    const int l0  = blockIdx.z * 256;

    // gate softmax (2 elems, every thread computes)
    float ga = gate[0], gb = gate[1];
    float gm = fmaxf(ga, gb);
    float e0 = __expf(ga - gm), e1 = __expf(gb - gm);
    float g0 = e0 / (e0 + e1), g1 = e1 / (e0 + e1);

    // load halo'd input slab (32 x 270)
    for (int idx = tid; idx < 32 * 270; idx += 256) {
        int c = idx / 270, j = idx % 270;
        int l = l0 - 7 + j;
        xs[c][j] = (l >= 0 && l < L_) ? x[(b * QKC_ + c) * L_ + l] : 0.0f;
    }
    // weights for this o-chunk
    for (int idx = tid; idx < 16 * 32; idx += 256)
        pw_s[idx / 32][idx % 32] = pw[(oc0 + idx / 32) * 32 + (idx % 32)];
    for (int idx = tid; idx < 16 * 3; idx += 256)
        dw3_s[idx / 3][idx % 3] = dw3[(oc0 + idx / 3) * 3 + (idx % 3)];
    for (int idx = tid; idx < 16 * 15; idx += 256)
        dw15_s[idx / 15][idx % 15] = dw15[(oc0 + idx / 15) * 15 + (idx % 15)];
    __syncthreads();

    // phase 1: pointwise conv into ps (16 x 270)
    for (int idx = tid; idx < 16 * 270; idx += 256) {
        int o = idx / 270, j = idx % 270;
        float s = 0.0f;
        #pragma unroll
        for (int c = 0; c < 32; ++c) s = fmaf(xs[c][j], pw_s[o][c], s);
        ps[o][j] = s;
    }
    __syncthreads();

    // phase 2: gated depthwise, store f16 [head][l][c]
    for (int idx = tid; idx < 16 * 256; idx += 256) {
        int o = idx / 256, i = idx % 256;
        float s3 = 0.0f, s15 = 0.0f;
        #pragma unroll
        for (int t = 0; t < 3; ++t)  s3  = fmaf(ps[o][i + 6 + t], dw3_s[o][t],  s3);
        #pragma unroll
        for (int t = 0; t < 15; ++t) s15 = fmaf(ps[o][i + t],     dw15_s[o][t], s15);
        float y = (g0 * s3 + g1 * s15) * scale;
        int oo = oc0 + o;
        int h  = oo >> 5, c = oo & 31;
        out[(((size_t)(b * H_ + h)) * L_ + (l0 + i)) * QKC_ + c] = (_Float16)y;
    }
}

// ---------------------------------------------------------------------------
// Encoder for V (v_c == 1): out[(b*8+h)][l], f32.  Grid: 16 blocks x 256 thr.
// ---------------------------------------------------------------------------
__global__ void __launch_bounds__(256)
encode_v_kernel(const float* __restrict__ x,      // [16][1][1024]
                const float* __restrict__ pw,     // [8]
                const float* __restrict__ dw3,    // [8][3]
                const float* __restrict__ dw15,   // [8][15]
                const float* __restrict__ gate,   // [2]
                float* __restrict__ out)          // [128][1024]
{
    __shared__ float xs[1040];   // j -> l = j - 7
    const int tid = threadIdx.x;
    const int b   = blockIdx.x;

    float ga = gate[0], gb = gate[1];
    float gm = fmaxf(ga, gb);
    float e0 = __expf(ga - gm), e1 = __expf(gb - gm);
    float g0 = e0 / (e0 + e1), g1 = e1 / (e0 + e1);

    for (int j = tid; j < 1038; j += 256) {
        int l = j - 7;
        xs[j] = (l >= 0 && l < L_) ? x[b * L_ + l] : 0.0f;
    }
    __syncthreads();

    for (int idx = tid; idx < H_ * L_; idx += 256) {
        int h = idx >> 10, i = idx & 1023;
        float s3 = 0.0f, s15 = 0.0f;
        #pragma unroll
        for (int t = 0; t < 3; ++t)  s3  = fmaf(xs[i + 6 + t], dw3[h * 3 + t],  s3);
        #pragma unroll
        for (int t = 0; t < 15; ++t) s15 = fmaf(xs[i + t],     dw15[h * 15 + t], s15);
        out[(b * H_ + h) * L_ + i] = (g0 * s3 + g1 * s15) * pw[h];
    }
}

// ---------------------------------------------------------------------------
// Attention per head: S = K_tile^T Q via v_wmma_f32_16x16x32_f16 (K dim = 32
// channels, one WMMA per 16x16 tile), row softmax over m, accumulate
// out[m] += (v[l]/sumexp_l) * exp(S[l][m]-max_l).
// Grid: 128 blocks (one per head), 512 threads = 16 waves.
// ---------------------------------------------------------------------------
__global__ void __launch_bounds__(512)
attn_kernel(const _Float16* __restrict__ q_enc,   // [128][1024][32]
            const _Float16* __restrict__ k_enc,   // [128][1024][32]
            const float*    __restrict__ v_enc,   // [128][1024]
            float*          __restrict__ head_out)// [128][1024]
{
    __shared__ float S[16 * 1024];     // 64 KB score slab for one l-tile
    __shared__ float out_acc[1024];
    __shared__ float rowmax[16];
    __shared__ float coeff[16];

    const int bh   = blockIdx.x;
    const int tid  = threadIdx.x;
    const int lane = tid & 31;
    const int w    = tid >> 5;          // wave id 0..15

    const _Float16* qh = q_enc + (size_t)bh * L_ * QKC_;
    const _Float16* kh = k_enc + (size_t)bh * L_ * QKC_;

    out_acc[tid]       = 0.0f;
    out_acc[tid + 512] = 0.0f;
    __syncthreads();

    for (int lt = 0; lt < 64; ++lt) {
        const int l_base = lt * 16;

        // ---- A operand: K l-tile as 16(l) x 32(c), contiguous k order (same as B):
        // lanes 0-15: row=lane,    K = 0..15 contiguous
        // lanes16-31: row=lane-16, K = 16..31 contiguous
        const int arow = l_base + (lane & 15);
        const int cb   = (lane < 16) ? 0 : 16;
        const _Float16* kp = kh + (size_t)arow * QKC_;
        v8h alo = *(const v8h*)(kp + cb);
        v8h ahi = *(const v8h*)(kp + cb + 8);
        v16h A = __builtin_shufflevector(alo, ahi,
                    0, 1, 2, 3, 4, 5, 6, 7, 8, 9, 10, 11, 12, 13, 14, 15);

        // ---- each wave covers 4 of the 64 m-tiles
        #pragma unroll
        for (int mt4 = 0; mt4 < 4; ++mt4) {
            const int m_base = (w * 4 + mt4) * 16;
            // B operand: 32(c) x 16(m): lanes 0-15 col=lane, K=0..15 contiguous;
            // lanes 16-31 col=lane-16, K=16..31 contiguous.
            const int col = m_base + (lane & 15);
            const int qcb = (lane < 16) ? 0 : 16;
            v16h Bm = *(const v16h*)(qh + (size_t)col * QKC_ + qcb);

            v8f C = {};
            C = __builtin_amdgcn_wmma_f32_16x16x32_f16(
                    false, A, false, Bm, (short)0, C, false, false);

            // D layout: lanes 0-15: n=lane, VGPR r -> M=r ; lanes16-31: M=8+r
            const int n  = lane & 15;
            const int ro = (lane < 16) ? 0 : 8;
            #pragma unroll
            for (int r = 0; r < 8; ++r)
                S[(ro + r) * 1024 + m_base + n] = C[r];
        }
        __syncthreads();

        // ---- row stats: wave w owns row w
        {
            const float* Sr = S + w * 1024;
            float mx = -3.0e38f;
            for (int j = lane; j < 1024; j += 32) mx = fmaxf(mx, Sr[j]);
            #pragma unroll
            for (int off = 16; off >= 1; off >>= 1)
                mx = fmaxf(mx, __shfl_xor(mx, off, 32));
            float se = 0.0f;
            for (int j = lane; j < 1024; j += 32) se += __expf(Sr[j] - mx);
            #pragma unroll
            for (int off = 16; off >= 1; off >>= 1)
                se += __shfl_xor(se, off, 32);
            if (lane == 0) {
                rowmax[w] = mx;
                coeff[w]  = v_enc[(size_t)bh * L_ + l_base + w] / se;
            }
        }
        __syncthreads();

        // ---- accumulate over the 16 key rows of this l-tile
        for (int m = tid; m < 1024; m += 512) {
            float a = 0.0f;
            #pragma unroll
            for (int l = 0; l < 16; ++l)
                a = fmaf(coeff[l], __expf(S[l * 1024 + m] - rowmax[l]), a);
            out_acc[m] += a;
        }
        __syncthreads();
    }

    for (int m = tid; m < 1024; m += 512)
        head_out[(size_t)bh * L_ + m] = out_acc[m];
}

// ---------------------------------------------------------------------------
// Unify heads: out[b][l] = sum_h head_out[b*8+h][l] * u_w[h] + u_b[0]
// ---------------------------------------------------------------------------
__global__ void __launch_bounds__(256)
unify_kernel(const float* __restrict__ head_out,  // [128][1024]
             const float* __restrict__ u_w,       // [1][8][1]
             const float* __restrict__ u_b,       // [1]
             float* __restrict__ out)              // [16][1][1024]
{
    int idx = blockIdx.x * 256 + threadIdx.x;
    if (idx >= B_ * L_) return;
    int b = idx >> 10, l = idx & 1023;
    float s = u_b[0];
    #pragma unroll
    for (int h = 0; h < H_; ++h)
        s = fmaf(head_out[(b * H_ + h) * L_ + l], u_w[h], s);
    out[b * L_ + l] = s;
}

// ---------------------------------------------------------------------------
extern "C" void kernel_launch(void* const* d_in, const int* in_sizes, int n_in,
                              void* d_out, int out_size, void* d_ws, size_t ws_size,
                              hipStream_t stream) {
    const float* queries = (const float*)d_in[0];
    const float* keys    = (const float*)d_in[1];
    const float* values  = (const float*)d_in[2];
    const float* q_pw    = (const float*)d_in[3];
    const float* q_dw3   = (const float*)d_in[4];
    const float* q_dw15  = (const float*)d_in[5];
    const float* q_gate  = (const float*)d_in[6];
    const float* k_pw    = (const float*)d_in[7];
    const float* k_dw3   = (const float*)d_in[8];
    const float* k_dw15  = (const float*)d_in[9];
    const float* k_gate  = (const float*)d_in[10];
    const float* v_pw    = (const float*)d_in[11];
    const float* v_dw3   = (const float*)d_in[12];
    const float* v_dw15  = (const float*)d_in[13];
    const float* v_gate  = (const float*)d_in[14];
    const float* u_w     = (const float*)d_in[15];
    const float* u_b     = (const float*)d_in[16];

    // workspace carve-up
    char* ws = (char*)d_ws;
    _Float16* q_enc = (_Float16*)ws;                                // 8 MB
    _Float16* k_enc = (_Float16*)(ws + (size_t)8 * 1024 * 1024);    // 8 MB
    float* v_enc    = (float*)(ws + (size_t)16 * 1024 * 1024);      // 512 KB
    float* head_out = (float*)(ws + (size_t)16 * 1024 * 1024 + 512 * 1024);

    const float q_scale = 0.42044820762685725f;  // 32^-0.25
    const float k_scale = 1.0f;                  // 1^-0.25

    dim3 egrid(B_, 16, 4);
    encode_qk_kernel<<<egrid, 256, 0, stream>>>(queries, q_pw, q_dw3, q_dw15,
                                                q_gate, q_scale, q_enc);
    encode_qk_kernel<<<egrid, 256, 0, stream>>>(keys, k_pw, k_dw3, k_dw15,
                                                k_gate, k_scale, k_enc);
    encode_v_kernel<<<B_, 256, 0, stream>>>(values, v_pw, v_dw3, v_dw15,
                                            v_gate, v_enc);
    attn_kernel<<<NH_, 512, 0, stream>>>(q_enc, k_enc, v_enc, head_out);
    unify_kernel<<<(B_ * L_ + 255) / 256, 256, 0, stream>>>(head_out, u_w, u_b,
                                                            (float*)d_out);
}
